// QuantileNetwork_65017214926885
// MI455X (gfx1250) — matrix operand order repaired and checked by reading, not compiled, not run on hardware
//
#include <hip/hip_runtime.h>
#include <hip/hip_bf16.h>

#define BSZ  16384
#define SDIM 512
#define ADIM 64
#define XDIM 576
#define HID  1024
#define NQ   64
#define QED  64
#define EPSF 1e-5f
#define MT   32      // rows per block

typedef __attribute__((ext_vector_type(16))) _Float16 v16h;
typedef __attribute__((ext_vector_type(8)))  _Float16 v8h;
typedef __attribute__((ext_vector_type(8)))  float    v8f;

// ---------------- prep kernels ----------------

__global__ void pack_x_kernel(const float* __restrict__ state,
                              const float* __restrict__ action,
                              _Float16* __restrict__ Xh) {
  size_t idx = (size_t)blockIdx.x * blockDim.x + threadIdx.x;
  if (idx >= (size_t)BSZ * XDIM) return;
  int b = (int)(idx / XDIM);
  int k = (int)(idx - (size_t)b * XDIM);
  float v = (k < SDIM) ? state[(size_t)b * SDIM + k]
                       : action[(size_t)b * ADIM + (k - SDIM)];
  Xh[idx] = (_Float16)v;
}

template<int K, int N>
__global__ void transpose_f16_kernel(const float* __restrict__ W,
                                     _Float16* __restrict__ Wt) {
  size_t idx = (size_t)blockIdx.x * blockDim.x + threadIdx.x;
  if (idx >= (size_t)K * N) return;
  int n = (int)(idx / K);
  int k = (int)(idx - (size_t)n * K);
  Wt[idx] = (_Float16)W[(size_t)k * N + n];   // Wt is N x K row-major
}

// tvec[q] = (relu(tau_q*Wq1+bq1) @ Wq2 + bq2) @ Wt + bh
// restructured: cj[j] = sum_k Wq2[j,k]*Wt[k];  c0 = sum_k bq2[k]*Wt[k]
__global__ void tvec_kernel(const float* __restrict__ Wq1, const float* __restrict__ bq1,
                            const float* __restrict__ Wq2, const float* __restrict__ bq2,
                            const float* __restrict__ Wh,  const float* __restrict__ bh,
                            float* __restrict__ tvec) {
  __shared__ float cj[QED];
  __shared__ float c0s;
  int t = threadIdx.x;              // 64 threads
  float cc = 0.0f;
  for (int k = 0; k < QED; ++k) cc += Wq2[t * QED + k] * Wh[HID + k];
  cj[t] = cc;
  if (t == 0) {
    float c = 0.0f;
    for (int k = 0; k < QED; ++k) c += bq2[k] * Wh[HID + k];
    c0s = c;
  }
  __syncthreads();
  float tau = ((float)t + 0.5f) / (float)NQ;
  float acc = c0s;
  for (int j = 0; j < QED; ++j)
    acc += fmaxf(tau * Wq1[j] + bq1[j], 0.0f) * cj[j];
  tvec[t] = acc + bh[0];
}

// ---------------- fused GEMM + bias + LayerNorm + ReLU (+ final row-dot) ----------------
// Block: 256 threads (8 waves). Tile: MT(=32) rows x HID(=1024) cols.
// Wave w owns columns [w*128, w*128+128): 2 m-tiles x 8 n-tiles of 16x16 WMMA.

template<int KDIM, bool FINAL>
__global__ __launch_bounds__(256)
void fused_layer_kernel(const _Float16* __restrict__ A,   // BSZ x KDIM (row major f16)
                        const _Float16* __restrict__ Bt,  // HID x KDIM (W^T, f16)
                        const float* __restrict__ bias,
                        const float* __restrict__ gamma,
                        const float* __restrict__ beta,
                        const float* __restrict__ Wf,     // HID        (FINAL only)
                        const float* __restrict__ tvec,   // NQ         (FINAL only)
                        _Float16* __restrict__ Hout,      // BSZ x HID  (!FINAL)
                        float* __restrict__ Qout)         // BSZ x NQ   (FINAL)
{
  constexpr int ASTR = KDIM + 8;               // halves; pad for LDS bank spread
  __shared__ float lds_c[MT * HID];            // 128 KB; aliased as f16 A tile in K-loop
  _Float16* lds_a = (_Float16*)lds_c;

  const int tid  = threadIdx.x;
  const int wave = tid >> 5;
  const int lane = tid & 31;
  const int lrow = lane & 15;
  const int kblk = lane >> 4;
  const size_t row0 = (size_t)blockIdx.x * MT;
  const int nbase = wave * 128;

  // ---- stage A tile (MT x KDIM, f16) into LDS via async global->LDS DMA ----
  // Per-lane 16B copies let us keep the padded (KDIM+8) LDS stride for
  // conflict-free ds_load_b128 fragment reads. Tracked by ASYNCcnt.
  {
    constexpr int nchunk = MT * KDIM / 8;      // multiples of 256 -> EXEC always full
    for (int c = tid; c < nchunk; c += 256) {
      int off = c * 8;
      int r = off / KDIM;
      int k = off - r * KDIM;
      unsigned ldsoff = (unsigned)(uintptr_t)(lds_a + r * ASTR + k);
      const _Float16* gp = A + (row0 + r) * KDIM + k;
      asm volatile("global_load_async_to_lds_b128 %0, %1, off"
                   :: "v"(ldsoff), "v"(gp) : "memory");
    }
    asm volatile("s_wait_asynccnt 0" ::: "memory");
  }
  __syncthreads();

  // ---- K loop: D = A x B (fp16 in, fp32 acc) ----
  v8f acc[2][8] = {};
  for (int k0 = 0; k0 < KDIM; k0 += 32) {
    // A fragment: lane m=lrow, element e:
    //   K = kblk*16 + e. Two contiguous 16B LDS loads per m-tile.
    v16h afrag[2];
#pragma unroll
    for (int mt = 0; mt < 2; ++mt) {
      const _Float16* ap = lds_a + (lrow + 16 * mt) * ASTR + k0 + kblk * 16;
      v8h alo = *(const v8h*)ap;
      v8h ahi = *(const v8h*)(ap + 8);
      afrag[mt] = __builtin_shufflevector(alo, ahi,
                    0,1,2,3,4,5,6,7,8,9,10,11,12,13,14,15);
    }
#pragma unroll
    for (int nt = 0; nt < 8; ++nt) {
      // B fragment: lane n=lrow, K = k0 + 16*kblk + e -> contiguous 32B in W^T
      const _Float16* bp = Bt + (size_t)(nbase + nt * 16 + lrow) * KDIM
                              + k0 + kblk * 16;
      v16h bfrag = *(const v16h*)bp;
#pragma unroll
      for (int mt = 0; mt < 2; ++mt)
        acc[mt][nt] = __builtin_amdgcn_wmma_f32_16x16x32_f16(
            false, afrag[mt], false, bfrag, (short)0, acc[mt][nt], false, false);
    }
  }
  __syncthreads();   // done with lds_a; reuse as fp32 C

  // ---- C + bias -> LDS (MT x HID fp32). C layout: VGPR r -> M = r + 8*kblk, N = lrow.
#pragma unroll
  for (int nt = 0; nt < 8; ++nt) {
    int n = nbase + nt * 16 + lrow;
    float bv = bias[n];
#pragma unroll
    for (int mt = 0; mt < 2; ++mt)
#pragma unroll
      for (int r = 0; r < 8; ++r)
        lds_c[(mt * 16 + kblk * 8 + r) * HID + n] = acc[mt][nt][r] + bv;
  }
  __syncthreads();

  // ---- LayerNorm + ReLU; each wave normalizes 4 rows ----
  float rdot[4];
#pragma unroll
  for (int rr = 0; rr < 4; ++rr) {
    int row = wave * 4 + rr;
    float s = 0.0f, ss = 0.0f;
#pragma unroll
    for (int i = 0; i < HID / 32; ++i) {
      float v = lds_c[row * HID + i * 32 + lane];
      s += v; ss += v * v;
    }
#pragma unroll
    for (int off = 16; off > 0; off >>= 1) {
      s  += __shfl_xor(s,  off, 32);
      ss += __shfl_xor(ss, off, 32);
    }
    float mu  = s * (1.0f / HID);
    float var = ss * (1.0f / HID) - mu * mu;
    float rs  = rsqrtf(var + EPSF);
    float aq = 0.0f;
#pragma unroll
    for (int i = 0; i < HID / 32; ++i) {
      int n = i * 32 + lane;
      float v = lds_c[row * HID + n];
      float h = fmaxf((v - mu) * rs * gamma[n] + beta[n], 0.0f);
      if (FINAL) aq += h * Wf[n];
      else       Hout[(row0 + row) * HID + n] = (_Float16)h;
    }
    if (FINAL) {
#pragma unroll
      for (int off = 16; off > 0; off >>= 1) aq += __shfl_xor(aq, off, 32);
      rdot[rr] = aq;
    }
  }

  if (FINAL) {
    __syncthreads();                       // all rows read before reuse
    if (lane == 0) {
#pragma unroll
      for (int rr = 0; rr < 4; ++rr) lds_c[wave * 4 + rr] = rdot[rr];
    }
    __syncthreads();
    for (int j = tid; j < MT * NQ; j += 256) {
      int r = j >> 6;
      int q = j & 63;
      Qout[(row0 + r) * NQ + q] = lds_c[r] + tvec[q];
    }
  }
}

// ---------------- launcher ----------------

extern "C" void kernel_launch(void* const* d_in, const int* in_sizes, int n_in,
                              void* d_out, int out_size, void* d_ws, size_t ws_size,
                              hipStream_t stream) {
  (void)in_sizes; (void)n_in; (void)out_size; (void)ws_size;
  const float* state  = (const float*)d_in[0];
  const float* action = (const float*)d_in[1];
  const float* W1  = (const float*)d_in[2];
  const float* b1  = (const float*)d_in[3];
  const float* g1  = (const float*)d_in[4];
  const float* be1 = (const float*)d_in[5];
  const float* W2  = (const float*)d_in[6];
  const float* b2  = (const float*)d_in[7];
  const float* g2  = (const float*)d_in[8];
  const float* be2 = (const float*)d_in[9];
  const float* Wq1 = (const float*)d_in[10];
  const float* bq1 = (const float*)d_in[11];
  const float* Wq2 = (const float*)d_in[12];
  const float* bq2 = (const float*)d_in[13];
  const float* Wh  = (const float*)d_in[14];
  const float* bh  = (const float*)d_in[15];
  float* Qout = (float*)d_out;

  // workspace layout (all offsets 32B-aligned): ~53.1 MB total
  char* ws = (char*)d_ws;
  _Float16* Xh   = (_Float16*)ws;                       // BSZ*XDIM
  _Float16* H1   = Xh  + (size_t)BSZ * XDIM;            // BSZ*HID
  _Float16* W1t  = H1  + (size_t)BSZ * HID;             // HID*XDIM
  _Float16* W2t  = W1t + (size_t)HID * XDIM;            // HID*HID
  float*    tvec = (float*)(W2t + (size_t)HID * HID);   // NQ

  pack_x_kernel<<<(BSZ * XDIM + 255) / 256, 256, 0, stream>>>(state, action, Xh);
  transpose_f16_kernel<XDIM, HID><<<(XDIM * HID + 255) / 256, 256, 0, stream>>>(W1, W1t);
  transpose_f16_kernel<HID,  HID><<<(HID  * HID + 255) / 256, 256, 0, stream>>>(W2, W2t);
  tvec_kernel<<<1, NQ, 0, stream>>>(Wq1, bq1, Wq2, bq2, Wh, bh, tvec);

  fused_layer_kernel<XDIM, false><<<BSZ / MT, 256, 0, stream>>>(
      Xh, W1t, b1, g1, be1, nullptr, nullptr, H1, nullptr);
  fused_layer_kernel<HID, true><<<BSZ / MT, 256, 0, stream>>>(
      H1, W2t, b2, g2, be2, /*Wf=*/Wh, tvec, nullptr, Qout);
}
